// ACFPN_78950088835815
// MI455X (gfx1250) — matrix operand order repaired and checked by reading, not compiled, not run on hardware
//
#include <hip/hip_runtime.h>

// ---------------------------------------------------------------------------
// Types for CDNA5 WMMA
// ---------------------------------------------------------------------------
typedef __bf16 bf16_t;
typedef __attribute__((ext_vector_type(16))) __bf16 v16bf;
typedef __attribute__((ext_vector_type(8)))  __bf16 v8bf;
typedef __attribute__((ext_vector_type(4)))  __bf16 v4bf;
typedef __attribute__((ext_vector_type(8)))  float  v8f;

#define KSTEP 32
#define NB 2          // batch size (fixed by reference)

// ---------------------------------------------------------------------------
// f32 -> bf16 tensor conversion (weights are converted once, reused by all
// conv tile blocks; removes per-tile cvt work and halves weight traffic)
// ---------------------------------------------------------------------------
__global__ void cvtbf16_kernel(const float* __restrict__ src, bf16_t* __restrict__ dst, int n)
{
    int i = blockIdx.x * blockDim.x + threadIdx.x;
    if (i < n) dst[i] = (bf16_t)src[i];
}

// ---------------------------------------------------------------------------
// Implicit-GEMM conv kernel using v_wmma_f32_16x16x32_bf16.
//   Out[b, co, y, x] = sum_{ci,r,s} W[co,ci,r,s] * X[b, ci, y+r*dil-pad, x+s*dil-pad]
// GEMM view: M = Cout, N = B*H*W, K = Cin*KH*KW (im2col in address math).
// Block = 256 threads = 8 waves; macro tile = 128 (M) x 64 (N);
// 4 WMMA per wave per K-step sharing one A fragment.
// A-tile staging uses gfx1250 async global->LDS b128 copies (ASYNCcnt),
// overlapping with the B-side im2col loads; no VGPR round trip for weights.
// Templated on KHW (1 or 9): 1x1 path has no im2col math at all; 3x3 path
// uses constant-divisor mul-shift instead of runtime integer division.
// Weights arrive pre-converted to bf16 ([Cout][K], row-major).
// LDS rows padded to 40 elems (80 B) so all fragment reads are aligned b128.
// Xcs/Ocs/Rcs are per-batch channel strides (lets us read/write concat slices).
// ---------------------------------------------------------------------------
template <int KHW_T>
__global__ __launch_bounds__(256) void conv_wmma_kernel(
    const float* __restrict__ X, const bf16_t* __restrict__ Wb,
    const float* __restrict__ Bias, float* __restrict__ Out,
    const float* __restrict__ Res,
    int Cin, int H, int W, int Cout,
    int pad, int dil,
    int Xcs, int Ocs, int Rcs, int relu)
{
    const int KW   = (KHW_T == 9) ? 3 : 1;
    const int K    = Cin * KHW_T;
    const int HWp  = H * W;
    const int Npix = NB * HWp;

    const int n0   = blockIdx.x * 64;
    const int co0  = blockIdx.y * 128;
    const int tid  = threadIdx.x;
    const int lane = tid & 31;
    const int wave = tid >> 5;
    const int half = (lane >> 4) & 1;
    const int l16  = lane & 15;

    __shared__ __align__(16) bf16_t As [128][40];   // [m][k]   (row stride 80 B)
    __shared__ __align__(16) bf16_t BsT[64][40];    // [n][k]   (transposed!)

    v8f acc[4] = {{}, {}, {}, {}};

    // ---- B staging: lane = pixel (coalesced), wave = k-quad; 2 pixels/thread ----
    const int bkk0 = wave * 4;      // k offset 0,4,...,28
    int  pb[2], py[2], px[2], phw[2];
    bool pok[2];
    #pragma unroll
    for (int p = 0; p < 2; ++p) {
        int nn = n0 + p * 32 + lane;
        pok[p] = nn < Npix;
        int b = 0, hw = 0;
        if (pok[p]) { b = nn / HWp; hw = nn - b * HWp; }
        pb[p] = b; phw[p] = hw;
        py[p] = hw / W;
        px[p] = hw - py[p] * W;
    }

    for (int k0 = 0; k0 < K; k0 += KSTEP) {
        // ---- stage A (bf16 weights) 128x32 via async global->LDS b128 DMA ----
        #pragma unroll
        for (int j = 0; j < 2; ++j) {
            int e8  = j * 256 + tid;        // 0..511 v8bf chunks
            int row = e8 >> 2;              // 0..127
            int kk  = (e8 & 3) * 8;         // 0,8,16,24
            int co  = co0 + row;
            if (co < Cout) {
                unsigned      ldsa = (unsigned)(size_t)&As[row][kk];
                const bf16_t* gsrc = &Wb[(size_t)co * K + k0 + kk];
                asm volatile("global_load_async_to_lds_b128 %0, %1, off"
                             :: "v"(ldsa), "v"(gsrc)
                             : "memory");
            } else {
                *(v8bf*)&As[row][kk] = (v8bf){};
            }
        }
        // ---- stage B (im2col activations) 32(K) x 64(N), transposed ----
        #pragma unroll
        for (int p = 0; p < 2; ++p) {
            v4bf b4;
            #pragma unroll
            for (int u = 0; u < 4; ++u) {
                int k = k0 + bkk0 + u;
                float v = 0.f;
                if (KHW_T == 1) {
                    if (pok[p] && k < K)
                        v = X[((size_t)pb[p] * Xcs + k) * HWp + phw[p]];
                } else {
                    if (pok[p] && k < K) {
                        int ci = k / KHW_T;          // constant divisor
                        int rs = k - ci * KHW_T;
                        int r  = rs / KW;
                        int s  = rs - r * KW;
                        int iy = py[p] + r * dil - pad;
                        int ix = px[p] + s * dil - pad;
                        if (iy >= 0 && iy < H && ix >= 0 && ix < W)
                            v = X[((size_t)pb[p] * Xcs + ci) * HWp + (size_t)iy * W + ix];
                    }
                }
                b4[u] = (bf16_t)v;
            }
            *(v4bf*)&BsT[p * 32 + lane][bkk0] = b4;
        }
        // wait for this wave's async LDS copies, then barrier with other waves
        asm volatile("s_wait_asynccnt 0x0" ::: "memory");
        __syncthreads();

        // ---- build fragments: all reads are aligned 16-byte LDS loads ----
        // A (16x32 16-bit, ISA 7.12.2): e<8 -> K = half*8+e ; e>=8 -> K = 16+half*8+(e-8)
        const int arow = wave * 16 + l16;
        v8bf a0 = *(const v8bf*)&As[arow][half * 8];
        v8bf a1 = *(const v8bf*)&As[arow][16 + half * 8];
        v16bf afrag = __builtin_shufflevector(a0, a1,
            0, 1, 2, 3, 4, 5, 6, 7, 8, 9, 10, 11, 12, 13, 14, 15);
        // B (32x16 16-bit): same K order as A: e<8 -> K = half*8+e ; e>=8 -> K = 16+half*8+(e-8)
        #pragma unroll
        for (int s = 0; s < 4; ++s) {
            v8bf b0 = *(const v8bf*)&BsT[s * 16 + l16][half * 8];
            v8bf b1 = *(const v8bf*)&BsT[s * 16 + l16][16 + half * 8];
            v16bf bfrag = __builtin_shufflevector(b0, b1,
                0, 1, 2, 3, 4, 5, 6, 7, 8, 9, 10, 11, 12, 13, 14, 15);
            acc[s] = __builtin_amdgcn_wmma_f32_16x16x32_bf16(
                false, afrag, false, bfrag, (short)0, acc[s], false, false);
        }
        __syncthreads();
    }

    // ---- epilogue: bias (+residual) (+relu) for the four N-subtiles ----
    #pragma unroll
    for (int s = 0; s < 4; ++s) {
        int n = n0 + s * 16 + l16;
        if (n < Npix) {
            int b  = n / HWp;
            int hw = n - b * HWp;
            #pragma unroll
            for (int r = 0; r < 8; ++r) {
                int m = co0 + wave * 16 + ((lane < 16) ? r : r + 8);
                if (m < Cout) {
                    float v = acc[s][r] + Bias[m];
                    if (Res)  v += Res[((size_t)b * Rcs + m) * HWp + hw];
                    if (relu) v = fmaxf(v, 0.f);
                    Out[((size_t)b * Ocs + m) * HWp + hw] = v;
                }
            }
        }
    }
}

// ---------------------------------------------------------------------------
// GroupNorm (32 groups) with optional fused ReLU; in-place safe.
// grid = (groups, B), block = 256
// ---------------------------------------------------------------------------
__global__ __launch_bounds__(256) void groupnorm_kernel(
    const float* __restrict__ x, float* __restrict__ y,
    const float* __restrict__ gamma, const float* __restrict__ beta,
    int C, int HW, int groups, int relu)
{
    int g = blockIdx.x, b = blockIdx.y;
    int cg = C / groups;
    int n  = cg * HW;
    const float* xp = x + ((size_t)b * C + (size_t)g * cg) * HW;
    float*       yp = y + ((size_t)b * C + (size_t)g * cg) * HW;

    float s = 0.f, s2 = 0.f;
    for (int i = threadIdx.x; i < n; i += 256) {
        float v = xp[i];
        s += v; s2 += v * v;
    }
    __shared__ float sh1[256], sh2[256];
    sh1[threadIdx.x] = s; sh2[threadIdx.x] = s2;
    __syncthreads();
    for (int off = 128; off > 0; off >>= 1) {
        if (threadIdx.x < off) {
            sh1[threadIdx.x] += sh1[threadIdx.x + off];
            sh2[threadIdx.x] += sh2[threadIdx.x + off];
        }
        __syncthreads();
    }
    float mean = sh1[0] / (float)n;
    float var  = sh2[0] / (float)n - mean * mean;
    float inv  = rsqrtf(var + 1e-5f);
    for (int i = threadIdx.x; i < n; i += 256) {
        int c = g * cg + i / HW;
        float v = (xp[i] - mean) * inv * gamma[c] + beta[c];
        if (relu) v = fmaxf(v, 0.f);
        yp[i] = v;
    }
}

// ---------------------------------------------------------------------------
// Criss-cross attention: energies + softmax over (H + W) keys per pixel.
// One thread per (b,h,w). Cq = 16. Diagonal of the H-branch masked to -inf.
// ---------------------------------------------------------------------------
__global__ void cc_att_kernel(const float* __restrict__ q, const float* __restrict__ k,
                              float* __restrict__ att, int Cq, int H, int W)
{
    int idx = blockIdx.x * blockDim.x + threadIdx.x;
    int total = NB * H * W;
    if (idx >= total) return;
    int b  = idx / (H * W);
    int hw = idx - b * (H * W);
    int h  = hw / W, w = hw - h * W;

    float qv[16];
    for (int c = 0; c < Cq; ++c)
        qv[c] = q[((size_t)(b * Cq + c) * H + h) * W + w];

    float e[64];
    float mx = -3.0e38f;
    for (int p = 0; p < H; ++p) {
        float s = 0.f;
        for (int c = 0; c < Cq; ++c)
            s += qv[c] * k[((size_t)(b * Cq + c) * H + p) * W + w];
        if (p == h) s = -3.0e38f;
        e[p] = s;
        mx = fmaxf(mx, s);
    }
    for (int qq = 0; qq < W; ++qq) {
        float s = 0.f;
        for (int c = 0; c < Cq; ++c)
            s += qv[c] * k[((size_t)(b * Cq + c) * H + h) * W + qq];
        e[H + qq] = s;
        mx = fmaxf(mx, s);
    }
    float sum = 0.f;
    for (int j = 0; j < H + W; ++j) { float t = __expf(e[j] - mx); e[j] = t; sum += t; }
    float r = 1.f / sum;
    float* ap = att + (size_t)idx * (H + W);
    for (int j = 0; j < H + W; ++j) ap[j] = e[j] * r;
}

// ---------------------------------------------------------------------------
// Apply both criss-cross attentions fused:
//   out = g1*(oH1+oW1) + g2*(oH2+oW2) + 2*dense
// One thread per (b,c,h,w).
// ---------------------------------------------------------------------------
__global__ void cc_apply_kernel(
    const float* __restrict__ v1, const float* __restrict__ a1, const float* __restrict__ g1,
    const float* __restrict__ v2, const float* __restrict__ a2, const float* __restrict__ g2,
    const float* __restrict__ xres, float* __restrict__ out,
    int C, int H, int W)
{
    int idx = blockIdx.x * blockDim.x + threadIdx.x;
    int total = NB * C * H * W;
    if (idx >= total) return;
    int w = idx % W;
    int t = idx / W;
    int h = t % H; t /= H;
    int c = t % C;
    int b = t / C;

    const float* ap1 = a1 + ((size_t)(b * H + h) * W + w) * (H + W);
    const float* ap2 = a2 + ((size_t)(b * H + h) * W + w) * (H + W);
    float o1 = 0.f, o2 = 0.f;
    for (int p = 0; p < H; ++p) {
        size_t i = ((size_t)(b * C + c) * H + p) * W + w;
        o1 += ap1[p] * v1[i];
        o2 += ap2[p] * v2[i];
    }
    for (int qq = 0; qq < W; ++qq) {
        size_t i = ((size_t)(b * C + c) * H + h) * W + qq;
        o1 += ap1[H + qq] * v1[i];
        o2 += ap2[H + qq] * v2[i];
    }
    out[idx] = g1[0] * o1 + g2[0] * o2 + 2.f * xres[idx];
}

// ---------------------------------------------------------------------------
// In-place: lat[b,c,y,x] += prev[b,c,y/2,x/2]   (2x nearest upsample + add)
// ---------------------------------------------------------------------------
__global__ void upadd_kernel(float* __restrict__ lat, const float* __restrict__ prev,
                             int C, int H, int W)
{
    int idx = blockIdx.x * blockDim.x + threadIdx.x;
    int total = NB * C * H * W;
    if (idx >= total) return;
    int x = idx % W;
    int t = idx / W;
    int y = t % H; t /= H;
    int c = t % C;
    int b = t / C;
    lat[idx] += prev[((size_t)(b * C + c) * (H / 2) + (y >> 1)) * (W / 2) + (x >> 1)];
}

// ---------------------------------------------------------------------------
// Strided channel-slice copy (for zero-copy concat bookkeeping)
// ---------------------------------------------------------------------------
__global__ void copych_kernel(const float* __restrict__ src, float* __restrict__ dst,
                              int C, int HW, int scs, int dcs)
{
    int idx = blockIdx.x * blockDim.x + threadIdx.x;
    int total = NB * C * HW;
    if (idx >= total) return;
    int p = idx % HW;
    int t = idx / HW;
    int c = t % C;
    int b = t / C;
    dst[((size_t)b * dcs + c) * HW + p] = src[((size_t)b * scs + c) * HW + p];
}

// ---------------------------------------------------------------------------
// Host orchestration
// ---------------------------------------------------------------------------
extern "C" void kernel_launch(void* const* d_in, const int* in_sizes, int n_in,
                              void* d_out, int out_size, void* d_ws, size_t ws_size,
                              hipStream_t stream)
{
    (void)in_sizes; (void)n_in; (void)out_size; (void)ws_size;
    auto F = [&](int i) { return (const float*)d_in[i]; };

    // --- input indices (setup_inputs dict order, params flattened depth-first) ---
    const float* x0 = F(0);   // (2,256,200,200)
    const float* x1 = F(1);   // (2,512,100,100)
    const float* x2 = F(2);   // (2,1024,50,50)
    const float* x3 = F(3);   // (2,2048,25,25)
    // dense.aspp{i}: 4+6i..9+6i = w1,b1,gn_g,gn_b,w2,b2 ; then 34..37 = wout,bout,gn_g,gn_b
    // cc : 38..44 = wq,bq,wk,bk,wv,bv,gamma ; ccf: 45..51
    // fpn: 52+4*i = inner_w, inner_b, layer_w, layer_b

    float* out = (float*)d_out;
    const size_t OFF_P2 = 0, OFF_P3 = 10240000, OFF_P4 = 12800000, OFF_P5 = 13440000;

    // --- workspace bump allocator (f32 region, then bf16 weight region) ---
    float* ws = (float*)d_ws;
    size_t wo = 0;
    auto alloc = [&](size_t n) { float* p = ws + wo; wo += n; return p; };
    const int HW3 = 25 * 25;
    float* cur    = alloc((size_t)NB * 2560 * HW3);   // growing ASPP concat stack
    float* t512   = alloc((size_t)NB * 512 * HW3);
    float* outsb  = alloc((size_t)NB * 640 * HW3);    // concat of o0..o4
    float* draw   = alloc((size_t)NB * 128 * HW3);    // pre-GN dense
    float* dense  = alloc((size_t)NB * 128 * HW3);
    float* inner3 = alloc((size_t)NB * 128 * HW3);
    float* q1 = alloc((size_t)NB * 16 * HW3);
    float* k1 = alloc((size_t)NB * 16 * HW3);
    float* v1 = alloc((size_t)NB * 128 * HW3);
    float* q2 = alloc((size_t)NB * 16 * HW3);
    float* k2 = alloc((size_t)NB * 16 * HW3);
    float* v2 = alloc((size_t)NB * 128 * HW3);
    float* att1 = alloc((size_t)NB * HW3 * 50);
    float* att2 = alloc((size_t)NB * HW3 * 50);
    float* resl = alloc((size_t)NB * 128 * HW3);
    float* lat2 = alloc((size_t)NB * 128 * 50 * 50);
    float* lat1 = alloc((size_t)NB * 128 * 100 * 100);
    float* lat0 = alloc((size_t)NB * 128 * 200 * 200);

    // bf16 weight pool (16-byte aligned region after the f32 scratch)
    bf16_t* wpool = (bf16_t*)(ws + ((wo + 7) & ~(size_t)7));
    size_t  wpo   = 0;
    // convert weight tensor input idx (n elements) into the pool, return bf16 ptr
    auto cw = [&](int idx, size_t n) {
        bf16_t* p = wpool + wpo;
        wpo += (n + 7) & ~(size_t)7;       // keep 16B alignment
        cvtbf16_kernel<<<((int)n + 255) / 256, 256, 0, stream>>>(F(idx), p, (int)n);
        return p;
    };

    // --- conv launcher (dispatch on 1x1 vs 3x3) ---
    auto conv = [&](const float* Xp, int Xcs, int Cin,
                    const bf16_t* Wp, const float* Bp,
                    float* Op, int Ocs, int Hh, int Ww, int Cout,
                    int KH, int pad, int dil, bool relu,
                    const float* Resp, int Rcs) {
        int Npix = NB * Hh * Ww;
        dim3 grid((Npix + 63) / 64, (Cout + 127) / 128);
        if (KH == 1)
            conv_wmma_kernel<1><<<grid, 256, 0, stream>>>(
                Xp, Wp, Bp, Op, Resp, Cin, Hh, Ww, Cout, pad, dil, Xcs, Ocs, Rcs, relu ? 1 : 0);
        else
            conv_wmma_kernel<9><<<grid, 256, 0, stream>>>(
                Xp, Wp, Bp, Op, Resp, Cin, Hh, Ww, Cout, pad, dil, Xcs, Ocs, Rcs, relu ? 1 : 0);
    };
    auto gnorm = [&](const float* xp, float* yp, const float* g, const float* b,
                     int C, int HW, int relu) {
        groupnorm_kernel<<<dim3(32, NB), 256, 0, stream>>>(xp, yp, g, b, C, HW, 32, relu);
    };

    // ================= pre-convert all conv weights to bf16 (once) =================
    bf16_t* w1b[5]; bf16_t* w2b[5];
    for (int i = 0; i < 5; ++i) {
        int cin = 2048 + 128 * i;
        w1b[i] = cw(4 + 6 * i, (size_t)512 * cin);
        w2b[i] = cw(8 + 6 * i, (size_t)128 * 512 * 9);
    }
    bf16_t* woutb  = cw(34, (size_t)128 * 640);
    bf16_t* wq1b   = cw(38, (size_t)16 * 128);
    bf16_t* wk1b   = cw(40, (size_t)16 * 128);
    bf16_t* wv1b   = cw(42, (size_t)128 * 128);
    bf16_t* wq2b   = cw(45, (size_t)16 * 128);
    bf16_t* wk2b   = cw(47, (size_t)16 * 128);
    bf16_t* wv2b   = cw(49, (size_t)128 * 128);
    bf16_t* innb[4]; bf16_t* layb[4];
    const int incs[4] = {256, 512, 1024, 2048};
    for (int i = 0; i < 4; ++i) {
        innb[i] = cw(52 + 4 * i, (size_t)128 * incs[i]);
        layb[i] = cw(54 + 4 * i, (size_t)128 * 128 * 9);
    }

    // ================= Dense ASPP =================
    // place x3 at channel offset 512 of 'cur' (2560-ch buffer)
    {
        int total = NB * 2048 * HW3;
        copych_kernel<<<(total + 255) / 256, 256, 0, stream>>>(x3, cur + (size_t)512 * HW3,
                                                               2048, HW3, 2048, 2560);
    }
    const int rates[5] = {3, 6, 12, 18, 24};
    for (int i = 0; i < 5; ++i) {
        int cin  = 2048 + 128 * i;
        int coff = 512 - 128 * i;
        const float* b1  = F(5 + 6 * i);
        const float* gng = F(6 + 6 * i);
        const float* gnb = F(7 + 6 * i);
        const float* b2  = F(9 + 6 * i);
        conv(cur + (size_t)coff * HW3, 2560, cin, w1b[i], b1, t512, 512, 25, 25, 512,
             1, 0, 1, false, nullptr, 0);
        gnorm(t512, t512, gng, gnb, 512, HW3, 1);   // GN + ReLU (in place)
        if (i < 4) {
            float* op = cur + (size_t)(coff - 128) * HW3;  // prepend into concat stack
            conv(t512, 512, 512, w2b[i], b2, op, 2560, 25, 25, 128,
                 3, rates[i], rates[i], true, nullptr, 0);
            int total = NB * 128 * HW3;
            copych_kernel<<<(total + 255) / 256, 256, 0, stream>>>(
                op, outsb + (size_t)(128 * i) * HW3, 128, HW3, 2560, 640);
        } else {
            conv(t512, 512, 512, w2b[i], b2, outsb + (size_t)(128 * 4) * HW3, 640, 25, 25, 128,
                 3, rates[i], rates[i], true, nullptr, 0);
        }
    }
    conv(outsb, 640, 640, woutb, F(35), draw, 128, 25, 25, 128, 1, 0, 1, false, nullptr, 0);
    gnorm(draw, dense, F(36), F(37), 128, HW3, 0);

    // ================= inner3 (last_inner) =================
    conv(x3, 2048, 2048, innb[3], F(65), inner3, 128, 25, 25, 128, 1, 0, 1, false, nullptr, 0);

    // ================= criss-cross x2 =================
    conv(dense, 128, 128, wq1b, F(39), q1, 16, 25, 25, 16, 1, 0, 1, false, nullptr, 0);
    conv(dense, 128, 128, wk1b, F(41), k1, 16, 25, 25, 16, 1, 0, 1, false, nullptr, 0);
    conv(dense, 128, 128, wv1b, F(43), v1, 128, 25, 25, 128, 1, 0, 1, false, nullptr, 0);
    conv(dense, 128, 128, wq2b, F(46), q2, 16, 25, 25, 16, 1, 0, 1, false, nullptr, 0);
    conv(dense, 128, 128, wk2b, F(48), k2, 16, 25, 25, 16, 1, 0, 1, false, nullptr, 0);
    conv(inner3, 128, 128, wv2b, F(50), v2, 128, 25, 25, 128, 1, 0, 1, false, nullptr, 0);
    {
        int tp = NB * HW3;
        cc_att_kernel<<<(tp + 127) / 128, 128, 0, stream>>>(q1, k1, att1, 16, 25, 25);
        cc_att_kernel<<<(tp + 127) / 128, 128, 0, stream>>>(q2, k2, att2, 16, 25, 25);
        int ta = NB * 128 * HW3;
        cc_apply_kernel<<<(ta + 255) / 256, 256, 0, stream>>>(
            v1, att1, F(44), v2, att2, F(51), dense, resl, 128, 25, 25);
    }

    // p5 = result + conv3x3(inner3, layer3)
    conv(inner3, 128, 128, layb[3], F(67), out + OFF_P5, 128, 25, 25, 128,
         3, 1, 1, false, resl, 128);

    // ================= FPN top-down =================
    // idx = 2
    conv(x2, 1024, 1024, innb[2], F(61), lat2, 128, 50, 50, 128, 1, 0, 1, false, nullptr, 0);
    { int t = NB * 128 * 50 * 50;
      upadd_kernel<<<(t + 255) / 256, 256, 0, stream>>>(lat2, inner3, 128, 50, 50); }
    conv(lat2, 128, 128, layb[2], F(63), out + OFF_P4, 128, 50, 50, 128, 3, 1, 1, false, nullptr, 0);
    // idx = 1
    conv(x1, 512, 512, innb[1], F(57), lat1, 128, 100, 100, 128, 1, 0, 1, false, nullptr, 0);
    { int t = NB * 128 * 100 * 100;
      upadd_kernel<<<(t + 255) / 256, 256, 0, stream>>>(lat1, lat2, 128, 100, 100); }
    conv(lat1, 128, 128, layb[1], F(59), out + OFF_P3, 128, 100, 100, 128, 3, 1, 1, false, nullptr, 0);
    // idx = 0
    conv(x0, 256, 256, innb[0], F(53), lat0, 128, 200, 200, 128, 1, 0, 1, false, nullptr, 0);
    { int t = NB * 128 * 200 * 200;
      upadd_kernel<<<(t + 255) / 256, 256, 0, stream>>>(lat0, lat1, 128, 200, 200); }
    conv(lat0, 128, 128, layb[0], F(55), out + OFF_P2, 128, 200, 200, 128, 3, 1, 1, false, nullptr, 0);
}
